// TrendAwareAttention_38903813767333
// MI455X (gfx1250) — matrix operand order repaired and checked by reading, not compiled, not run on hardware
//
#include <hip/hip_runtime.h>
#include <hip/hip_bf16.h>
#include <stdint.h>

// ---------------------------------------------------------------------------
// TrendAwareAttention on MI455X (gfx1250), all matmuls via v_wmma_f32_16x16x32_bf16
//   B=8 N=64 T=128 D=256 H=8 DK=32, BN=512
// Pipeline:
//   k0: convert/transpose weights f32 -> bf16  (conv weights -> [tap][cout][cin])
//   k1: conv1d(k=3,'same') GEMM for Q and K    -> bf16 (BN,T,D)
//   k2: linear V                               -> bf16 (BN,H,DK,T)  (head-transposed)
//   k3: per-(bn,h) attention (QK^T, mask, softmax, P·V) -> bf16 (BN,T,D)
//   k4: output linear -> f32 d_out; A-tile staged via GLOBAL_LOAD_ASYNC_TO_LDS
// ---------------------------------------------------------------------------

#define TB   8
#define TN   64
#define TT   128
#define TD   256
#define TH   8
#define TDK  32
#define TBN  (TB * TN)          // 512
#define SCALE_F 0.17677669529663687f   // 1/sqrt(32)
#define NEG_F  (-1e9f)

typedef __bf16 bf16_t;
typedef __attribute__((ext_vector_type(16))) __bf16 v16bf;
typedef __attribute__((ext_vector_type(8)))  float  v8f;

__device__ __forceinline__ unsigned short f2bf(float f) {
  unsigned int u = __float_as_uint(f);
  unsigned int r = u + 0x7FFFu + ((u >> 16) & 1u);   // round-to-nearest-even
  return (unsigned short)(r >> 16);
}

__device__ __forceinline__ v16bf load_frag2(const unsigned short* p0,
                                            const unsigned short* p1) {
  union { v16bf v; uint4 q[2]; } u;
  u.q[0] = *(const uint4*)p0;   // 16 bytes: 8 bf16
  u.q[1] = *(const uint4*)p1;   // 16 bytes: 8 bf16
  return u.v;
}

__device__ __forceinline__ v8f wmma_bf16(v16bf a, v16bf b, v8f c) {
  return __builtin_amdgcn_wmma_f32_16x16x32_bf16(
      /*neg_a=*/false, a, /*neg_b=*/false, b,
      /*c_mod=*/(short)0, c, /*reuse_a=*/false, /*reuse_b=*/false);
}

// Async copy of 16 bytes global -> LDS, tracked by ASYNCcnt (CDNA5).
// ldsaddr = low 32 bits of generic shared pointer (HW uses addr[31:0] for LDS).
__device__ __forceinline__ void async_g2l_b128(unsigned int lds_addr,
                                               const void* gaddr) {
  asm volatile("global_load_async_to_lds_b128 %0, %1, off"
               :: "v"(lds_addr), "v"(gaddr) : "memory");
}
__device__ __forceinline__ void wait_asynccnt0() {
  asm volatile("s_wait_asynccnt 0x0" ::: "memory");
}

// -------------------- ws ushort regions (element offsets) --------------------
//   WQ [0, 196608)        : [tap][cout][cin]  bf16
//   WK [196608, 393216)
//   WV [393216, 458752)   : [out][in] bf16
//   WO [458752, 524288)
//   QB [524288, +16777216) bf16 (BN,T,D)
//   KB next 16777216
//   VT next 16777216      bf16 (BN,H,DK,T)
//   XB next 16777216      bf16 (BN,T,D)
#define OFF_WK 196608
#define OFF_WV 393216
#define OFF_WO 458752
#define OFF_QB 524288
#define OFF_KB (OFF_QB + 16777216)
#define OFF_VT (OFF_KB + 16777216)
#define OFF_XB (OFF_VT + 16777216)

__global__ __launch_bounds__(256)
void cvt_weights_kernel(const float* __restrict__ wq, const float* __restrict__ wk,
                        const float* __restrict__ wv, const float* __restrict__ wo,
                        unsigned short* __restrict__ W) {
  const int i = blockIdx.x * blockDim.x + threadIdx.x;
  const int CW = 3 * TD * TD;    // 196608
  const int LW = TD * TD;        // 65536
  if (i < CW) {
    int tap = i / LW, r = i % LW;
    int cout = r / TD, cin = r % TD;
    int src = cout * TD * 3 + cin * 3 + tap;   // (D, D, K) torch layout
    W[i]           = f2bf(wq[src]);
    W[OFF_WK + i]  = f2bf(wk[src]);
  }
  if (i < LW) {
    W[OFF_WV + i] = f2bf(wv[i]);   // (out,in) already k-contiguous
    W[OFF_WO + i] = f2bf(wo[i]);
  }
}

// -------------------- k1: conv1d('same') as 3-tap shifted GEMM --------------------
// grid (BN, T/16), block 256 (8 waves); wave w owns output cols [w*32, w*32+32)
template <int TAPS>
__global__ __launch_bounds__(256)
void conv_gemm_kernel(const float* __restrict__ X,          // (BN,T,D) f32
                      const unsigned short* __restrict__ Wt,// [TAPS][cout][cin] bf16
                      const float* __restrict__ bias,       // (D)
                      unsigned short* __restrict__ Y) {     // (BN,T,D) bf16
  const int bn = blockIdx.x;
  const int t0 = blockIdx.y * 16;
  const int tid = threadIdx.x;
  const int wave = tid >> 5, lane = tid & 31;
  const int lm = lane & 15, half = lane >> 4;

  constexpr int HALO = (TAPS == 3) ? 1 : 0;
  constexpr int ROWS = 16 + 2 * HALO;
  constexpr int LSTR = TD + 8;                 // 264 elems -> 528B rows (16B aligned)
  __shared__ unsigned short As[ROWS * LSTR];

  // stage + convert once per element (f32 -> bf16)
  for (int idx = tid; idx < ROWS * TD; idx += 256) {
    int r = idx / TD, c = idx % TD;
    int t = t0 - HALO + r;
    float v = (t >= 0 && t < TT) ? X[((size_t)bn * TT + t) * TD + c] : 0.0f;
    As[r * LSTR + c] = f2bf(v);
  }
  __syncthreads();

  const int n0 = wave * 32;
  v8f acc0 = {}, acc1 = {};
#pragma unroll
  for (int tap = 0; tap < TAPS; ++tap) {
    const unsigned short* Wtap = Wt + (size_t)tap * TD * TD;
#pragma unroll
    for (int kc = 0; kc < TD; kc += 32) {
      const unsigned short* ar = As + (lm + tap) * LSTR + kc;          // A row = lm (+halo shift)
      v16bf a = load_frag2(ar + half * 16, ar + half * 16 + 8);
      const unsigned short* b0 = Wtap + (size_t)(n0 + lm) * TD + kc + half * 16;
      const unsigned short* b1 = Wtap + (size_t)(n0 + 16 + lm) * TD + kc + half * 16;
      v16bf fb0 = load_frag2(b0, b0 + 8);
      v16bf fb1 = load_frag2(b1, b1 + 8);
      acc0 = wmma_bf16(a, fb0, acc0);
      acc1 = wmma_bf16(a, fb1, acc1);
    }
  }
  const float bia0 = bias[n0 + lm], bia1 = bias[n0 + 16 + lm];
#pragma unroll
  for (int v = 0; v < 8; ++v) {
    int m = v + half * 8;
    size_t row = ((size_t)bn * TT + (t0 + m)) * TD;
    Y[row + n0 + lm]      = f2bf(acc0[v] + bia0);
    Y[row + n0 + 16 + lm] = f2bf(acc1[v] + bia1);
  }
}

// -------------------- k2: V linear, output head-transposed (BN,H,DK,T) --------------------
__global__ __launch_bounds__(256)
void vproj_kernel(const float* __restrict__ X,           // (BN,T,D) f32 (value)
                  const unsigned short* __restrict__ Wt, // [out][in] bf16
                  const float* __restrict__ bias,
                  unsigned short* __restrict__ Vt) {     // (BN,H,DK,T) bf16
  const int bn = blockIdx.x;
  const int t0 = blockIdx.y * 16;
  const int tid = threadIdx.x;
  const int wave = tid >> 5, lane = tid & 31;
  const int lm = lane & 15, half = lane >> 4;
  constexpr int LSTR = TD + 8;
  __shared__ unsigned short As[16 * LSTR];

  for (int idx = tid; idx < 16 * TD; idx += 256) {
    int r = idx / TD, c = idx % TD;
    As[r * LSTR + c] = f2bf(X[((size_t)bn * TT + t0 + r) * TD + c]);
  }
  __syncthreads();

  const int n0 = wave * 32;
  v8f acc0 = {}, acc1 = {};
#pragma unroll
  for (int kc = 0; kc < TD; kc += 32) {
    const unsigned short* ar = As + lm * LSTR + kc;
    v16bf a = load_frag2(ar + half * 16, ar + half * 16 + 8);
    const unsigned short* b0 = Wt + (size_t)(n0 + lm) * TD + kc + half * 16;
    const unsigned short* b1 = Wt + (size_t)(n0 + 16 + lm) * TD + kc + half * 16;
    acc0 = wmma_bf16(a, load_frag2(b0, b0 + 8), acc0);
    acc1 = wmma_bf16(a, load_frag2(b1, b1 + 8), acc1);
  }
  // store: out col n = h*32+dk  ->  Vt[((bn*H+h)*DK+dk)*T + t], 8 consecutive t per lane
  const float bia0 = bias[n0 + lm], bia1 = bias[n0 + 16 + lm];
  {
    int n = n0 + lm, h = n >> 5, dk = n & 31;
    union { uint4 q; unsigned short s[8]; } pk;
#pragma unroll
    for (int v = 0; v < 8; ++v) pk.s[v] = f2bf(acc0[v] + bia0);
    *(uint4*)&Vt[(((size_t)bn * TH + h) * TDK + dk) * TT + t0 + half * 8] = pk.q;
  }
  {
    int n = n0 + 16 + lm, h = n >> 5, dk = n & 31;
    union { uint4 q; unsigned short s[8]; } pk;
#pragma unroll
    for (int v = 0; v < 8; ++v) pk.s[v] = f2bf(acc1[v] + bia1);
    *(uint4*)&Vt[(((size_t)bn * TH + h) * TDK + dk) * TT + t0 + half * 8] = pk.q;
  }
}

// -------------------- k3: attention per (bn, h) --------------------
// grid (BN, H), block 256; wave w owns query rows [w*16, w*16+16)
__global__ __launch_bounds__(256)
void attn_kernel(const unsigned short* __restrict__ Qb,   // (BN,T,D) bf16
                 const unsigned short* __restrict__ Kb,   // (BN,T,D) bf16
                 const unsigned short* __restrict__ Vt,   // (BN,H,DK,T) bf16
                 const unsigned char* __restrict__ kpm,   // (B,N,T) bool
                 const int* __restrict__ amask,           // (T,T) int32
                 unsigned short* __restrict__ Xb) {       // (BN,T,D) bf16
  const int bn = blockIdx.x;
  const int h  = blockIdx.y;
  const int wave = threadIdx.x >> 5, lane = threadIdx.x & 31;
  const int lm = lane & 15, half = lane >> 4;
  const int tq0 = wave * 16;

  constexpr int PSTR = 136;                      // 16x128 P tile, 16B-aligned rows
  __shared__ unsigned short Ps[8 * 16 * PSTR];   // per-wave regions
  unsigned short* pw = &Ps[wave * 16 * PSTR];

  // Q A-frag: rows tq0+lm, K-dim = DK = 32 (single WMMA K step)
  const unsigned short* qr = Qb + ((size_t)bn * TT + tq0 + lm) * TD + h * TDK;
  v16bf qa = load_frag2(qr + half * 16, qr + half * 16 + 8);

  // scores: 8 key tiles of 16, one wmma each
  v8f sc[8];
#pragma unroll
  for (int j = 0; j < 8; ++j) {
    const unsigned short* kr = Kb + ((size_t)bn * TT + j * 16 + lm) * TD + h * TDK + half * 16;
    v8f z = {};
    sc[j] = wmma_bf16(qa, load_frag2(kr, kr + 8), z);
  }

  // scale + masks (lane col tk = j*16+lm; row tq = tq0 + half*8 + v)
#pragma unroll
  for (int j = 0; j < 8; ++j) {
    int tk = j * 16 + lm;
    bool kp = kpm[(size_t)bn * TT + tk] != 0;
#pragma unroll
    for (int v = 0; v < 8; ++v) {
      int tq = tq0 + half * 8 + v;
      float s = sc[j][v] * SCALE_F;
      if (kp || amask[tq * TT + tk] == 0) s = NEG_F;
      sc[j][v] = s;
    }
  }

  // rowwise softmax: reduce over 8 regs + 16 lanes (xor shuffles stay in half)
#pragma unroll
  for (int v = 0; v < 8; ++v) {
    float mx = -3.4e38f;
#pragma unroll
    for (int j = 0; j < 8; ++j) mx = fmaxf(mx, sc[j][v]);
    for (int off = 1; off < 16; off <<= 1) mx = fmaxf(mx, __shfl_xor(mx, off, 32));
    float e[8], sum = 0.f;
#pragma unroll
    for (int j = 0; j < 8; ++j) { e[j] = __expf(sc[j][v] - mx); sum += e[j]; }
    for (int off = 1; off < 16; off <<= 1) sum += __shfl_xor(sum, off, 32);
    float inv = 1.0f / sum;
    int m = half * 8 + v;
#pragma unroll
    for (int j = 0; j < 8; ++j)
      pw[m * PSTR + j * 16 + lm] = f2bf(e[j] * inv);
  }
  __syncthreads();

  // out = P (16x128) x V (128x32): 4 K-chunks, 2 n-tiles
  v8f o0 = {}, o1 = {};
#pragma unroll
  for (int c = 0; c < 4; ++c) {
    const unsigned short* pr = pw + lm * PSTR + c * 32;
    v16bf pa = load_frag2(pr + half * 16, pr + half * 16 + 8);
    const unsigned short* v0 = Vt + (((size_t)bn * TH + h) * TDK + lm) * TT + c * 32 + half * 16;
    const unsigned short* v1 = Vt + (((size_t)bn * TH + h) * TDK + 16 + lm) * TT + c * 32 + half * 16;
    o0 = wmma_bf16(pa, load_frag2(v0, v0 + 8), o0);
    o1 = wmma_bf16(pa, load_frag2(v1, v1 + 8), o1);
  }
#pragma unroll
  for (int v = 0; v < 8; ++v) {
    int tq = tq0 + half * 8 + v;
    size_t row = ((size_t)bn * TT + tq) * TD + h * TDK;
    Xb[row + lm]      = f2bf(o0[v]);
    Xb[row + 16 + lm] = f2bf(o1[v]);
  }
}

// -------------------- k4: output linear -> f32 --------------------
// A tile is already bf16 in ws: stage it with CDNA5 async global->LDS DMA
// (no VGPR round-trip), then the usual WMMA GEMM.
__global__ __launch_bounds__(256)
void oproj_kernel(const unsigned short* __restrict__ Xb,  // (BN,T,D) bf16
                  const unsigned short* __restrict__ Wt,  // [out][in] bf16
                  const float* __restrict__ bias,
                  float* __restrict__ Out) {              // (B,N,T,D) f32
  const int bn = blockIdx.x;
  const int t0 = blockIdx.y * 16;
  const int tid = threadIdx.x;
  const int wave = tid >> 5, lane = tid & 31;
  const int lm = lane & 15, half = lane >> 4;
  constexpr int LSTR = TD + 8;
  __shared__ unsigned short As[16 * LSTR];

  // 16 rows x 512B = 512 x 16B chunks; 2 async b128 copies per thread
  {
    const unsigned short* src = Xb + ((size_t)bn * TT + t0) * TD;
#pragma unroll
    for (int it = 0; it < 2; ++it) {
      int e = tid + it * 256;          // chunk id 0..511
      int r = e >> 5;                  // row 0..15
      int c8 = e & 31;                 // 16B chunk in row (32 per row)
      unsigned int lds = (unsigned int)(size_t)(As + r * LSTR + c8 * 8);
      async_g2l_b128(lds, src + r * TD + c8 * 8);
    }
    wait_asynccnt0();
  }
  __syncthreads();

  const int n0 = wave * 32;
  v8f acc0 = {}, acc1 = {};
#pragma unroll
  for (int kc = 0; kc < TD; kc += 32) {
    const unsigned short* ar = As + lm * LSTR + kc;
    v16bf a = load_frag2(ar + half * 16, ar + half * 16 + 8);
    const unsigned short* b0 = Wt + (size_t)(n0 + lm) * TD + kc + half * 16;
    const unsigned short* b1 = Wt + (size_t)(n0 + 16 + lm) * TD + kc + half * 16;
    acc0 = wmma_bf16(a, load_frag2(b0, b0 + 8), acc0);
    acc1 = wmma_bf16(a, load_frag2(b1, b1 + 8), acc1);
  }
  const float bia0 = bias[n0 + lm], bia1 = bias[n0 + 16 + lm];
#pragma unroll
  for (int v = 0; v < 8; ++v) {
    int m = v + half * 8;
    size_t row = ((size_t)bn * TT + (t0 + m)) * TD;
    Out[row + n0 + lm]      = acc0[v] + bia0;
    Out[row + n0 + 16 + lm] = acc1[v] + bia1;
  }
}

// -------------------- host --------------------
extern "C" void kernel_launch(void* const* d_in, const int* in_sizes, int n_in,
                              void* d_out, int out_size, void* d_ws, size_t ws_size,
                              hipStream_t stream) {
  const float* query = (const float*)d_in[0];
  const float* key   = (const float*)d_in[1];
  const float* value = (const float*)d_in[2];
  const unsigned char* kpm = (const unsigned char*)d_in[3];   // bool (1 byte)
  const int*   amask = (const int*)d_in[4];
  const float* wq = (const float*)d_in[5];
  const float* bq = (const float*)d_in[6];
  const float* wk = (const float*)d_in[7];
  const float* bk = (const float*)d_in[8];
  const float* wv = (const float*)d_in[9];
  const float* bv = (const float*)d_in[10];
  const float* wo = (const float*)d_in[11];
  const float* bo = (const float*)d_in[12];

  unsigned short* W = (unsigned short*)d_ws;
  unsigned short* WQ = W;
  unsigned short* WK = W + OFF_WK;
  unsigned short* WV = W + OFF_WV;
  unsigned short* WO = W + OFF_WO;
  unsigned short* QB = W + OFF_QB;
  unsigned short* KB = W + OFF_KB;
  unsigned short* VT = W + OFF_VT;
  unsigned short* XB = W + OFF_XB;

  cvt_weights_kernel<<<768, 256, 0, stream>>>(wq, wk, wv, wo, W);

  dim3 grid(TBN, TT / 16);
  conv_gemm_kernel<3><<<grid, 256, 0, stream>>>(query, WQ, bq, QB);
  conv_gemm_kernel<3><<<grid, 256, 0, stream>>>(key,   WK, bk, KB);
  vproj_kernel<<<grid, 256, 0, stream>>>(value, WV, bv, VT);

  dim3 agrid(TBN, TH);
  attn_kernel<<<agrid, 256, 0, stream>>>(QB, KB, VT, kpm, amask, XB);

  oproj_kernel<<<grid, 256, 0, stream>>>(XB, WO, bo, (float*)d_out);
}
